// RNN_35708358099218
// MI455X (gfx1250) — matrix operand order repaired and checked by reading, not compiled, not run on hardware
//
#include <hip/hip_runtime.h>
#include <hip/hip_bf16.h>
#include <math.h>

// ---------------------------------------------------------------------------
// RNN bank (15 RNNs, H=256, input=2, T=64, B=512) on gfx1250.
// Wh[r] resident in LDS as f16 (transposed, +8-half row pad -> conflict-free
// ds_load_b128 WMMA fragments). Recurrent GEMM via v_wmma_f32_16x16x32_f16.
// Wave tiling 2M x 4N (2x B / 4x A LDS reuse). Ping-pong h tile -> single
// barrier per time step on the serial recurrence chain. Epilogue constants
// hoisted to registers; fast exact-identity tanh via v_exp_f32 + v_rcp_f32.
// Block = (RNN r, 64-row batch tile); RNN r starts at t=r (== ref masking).
// ---------------------------------------------------------------------------

typedef _Float16 v16h __attribute__((ext_vector_type(16)));
typedef _Float16 v8h  __attribute__((ext_vector_type(8)));
typedef float    v8f  __attribute__((ext_vector_type(8)));
typedef float    v2f  __attribute__((ext_vector_type(2)));

#define NUM_RNNS 15
#define HID      256
#define TSTEPS   64
#define BATCH    512
#define BTILE    64
#define NBT      (BATCH / BTILE)          // 8 batch tiles
#define WPAD     264                       // 256 + 8 halves: row stride 528B
#define HPAD     264

__device__ __forceinline__ float tanh_fast(float v) {
    // tanh(x) = 1 - 2/(exp(2x)+1); exp(2x) = exp2(x * 2/ln2).
    // Saturates correctly: x -> +inf => 1, x -> -inf => -1.
    const float e = __builtin_amdgcn_exp2f(v * 2.885390081777927f);
    const float r = __builtin_amdgcn_rcpf(e + 1.0f);
    return __builtin_fmaf(-2.0f, r, 1.0f);
}

__global__ __launch_bounds__(256) void rnn_bank_kernel(
    const float* __restrict__ x,     // [512][128]  (b, t*2+c)
    const float* __restrict__ Wx,    // [15][2][256]
    const float* __restrict__ Wh,    // [15][256][256]
    const float* __restrict__ bvec,  // [15][256]
    const float* __restrict__ fcw,   // [15*256][1]
    float* __restrict__ ws)          // [15][512] partial FC dot products
{
    __shared__ _Float16 whT [HID * WPAD];      // Wh[r]^T [n][k] f16  (132 KB)
    __shared__ _Float16 hbuf[2][BTILE * HPAD]; // ping-pong h tiles   ( 66 KB)
    __shared__ float    xall[BTILE * TSTEPS*2];// x tile [m][t*2+c]   ( 32 KB)
    __shared__ float    wxl [2 * HID];
    __shared__ float    bbl [HID];
    __shared__ float    fcwl[HID];
    __shared__ float    red [256];

    const int tid  = threadIdx.x;
    const int r    = blockIdx.x % NUM_RNNS;   // RNN index
    const int tb   = blockIdx.x / NUM_RNNS;   // batch tile
    const int b0   = tb * BTILE;

    const int wave = tid >> 5;
    const int lane = tid & 31;
    const int mg   = wave & 1;                // M group: rows mg*32 .. +31
    const int nq   = wave >> 1;               // N quad:  n tiles nq*4 .. +3

    // ---- one-time loads ----
    for (int i = tid; i < HID * HID; i += 256) {
        const int k = i >> 8;                 // h_in
        const int n = i & 255;                // h_out
        whT[n * WPAD + k] = (_Float16)Wh[r * HID * HID + i];
    }
    for (int i = tid; i < BTILE * TSTEPS * 2; i += 256)
        xall[i] = x[(size_t)b0 * (TSTEPS * 2) + i];
    for (int i = tid; i < 2 * HID; i += 256) wxl[i] = Wx[r * 2 * HID + i];
    if (tid < HID) {
        bbl [tid] = bvec[r * HID + tid];
        fcwl[tid] = fcw [r * HID + tid];
    }
    for (int i = tid; i < BTILE * HPAD; i += 256)
        hbuf[0][i] = (_Float16)0.0f;          // h_{-1} = 0 (read at t = r)

    // WMMA fragment addressing (ISA 7.12.2 layouts)
    const int arow  = lane & 15;              // A: M = arow
    const int akoff = (lane < 16) ? 0 : 8;    // A: lanes 16-31 start at K+8
    const int bcol  = lane & 15;              // B: N = bcol
    const int bkoff = (lane < 16) ? 0 : 8;    // B: lanes 16-31 start at K+8
    const int mrow  = (lane >> 4) << 3;       // C: lanes 16-31 -> M += 8
    const int nloc  = lane & 15;

    __syncthreads();

    // t-invariant epilogue constants -> registers (LDS reads once, not per t)
    float w0r[4], w1r[4], bnr[4];
    #pragma unroll
    for (int j = 0; j < 4; ++j) {
        const int n = (nq * 4 + j) * 16 + nloc;
        w0r[j] = wxl[n];
        w1r[j] = wxl[HID + n];
        bnr[j] = bbl[n];
    }

    // ---- recurrence: RNN r is identically zero until t >= r ----
    int cur = 0;                              // buffer holding h_{t-1}
    for (int t = r; t < TSTEPS; ++t) {
        const _Float16* __restrict__ hin  = hbuf[cur];
        _Float16*       __restrict__ hout = hbuf[cur ^ 1];

        // acc tiles: [i: M tile 0..1][j: N tile 0..3]
        v8f acc[2][4];
        const v8f vzero = {0.f, 0.f, 0.f, 0.f, 0.f, 0.f, 0.f, 0.f};
        #pragma unroll
        for (int i = 0; i < 2; ++i)
            #pragma unroll
            for (int j = 0; j < 4; ++j) acc[i][j] = vzero;

        #pragma unroll
        for (int ks = 0; ks < 8; ++ks) {
            const int k0 = ks * 32;
            // issue all fragment loads for this k-step up front (12x b128)
            v16h afrag[2], bfrag[4];
            #pragma unroll
            for (int i = 0; i < 2; ++i) {
                const _Float16* ap =
                    &hin[(mg * 32 + i * 16 + arow) * HPAD + k0 + akoff];
                ((v8h*)&afrag[i])[0] = *(const v8h*)(ap);       // K koff..+7
                ((v8h*)&afrag[i])[1] = *(const v8h*)(ap + 16);  // K koff+16..+23
            }
            #pragma unroll
            for (int j = 0; j < 4; ++j) {
                const _Float16* bp =
                    &whT[((nq * 4 + j) * 16 + bcol) * WPAD + k0 + bkoff];
                ((v8h*)&bfrag[j])[0] = *(const v8h*)(bp);       // K koff..+7
                ((v8h*)&bfrag[j])[1] = *(const v8h*)(bp + 16);  // K koff+16..+23
            }
            #pragma unroll
            for (int j = 0; j < 4; ++j)
                #pragma unroll
                for (int i = 0; i < 2; ++i)
                    acc[i][j] = __builtin_amdgcn_wmma_f32_16x16x32_f16(
                        false, afrag[i], false, bfrag[j], (short)0,
                        acc[i][j], false, false);
        }

        // ---- epilogue: h_t = tanh(acc + x_t@Wx + b) -> other buffer ----
        #pragma unroll
        for (int i = 0; i < 2; ++i) {
            #pragma unroll
            for (int v = 0; v < 8; ++v) {
                const int m = mg * 32 + i * 16 + mrow + v;
                const v2f xv = *(const v2f*)&xall[m * (TSTEPS * 2) + t * 2];
                #pragma unroll
                for (int j = 0; j < 4; ++j) {
                    const int n = (nq * 4 + j) * 16 + nloc;
                    const float val = acc[i][j][v] + bnr[j]
                                    + __builtin_fmaf(xv[0], w0r[j],
                                                     xv[1] * w1r[j]);
                    hout[m * HPAD + n] = (_Float16)tanh_fast(val);
                }
            }
        }

        __syncthreads();   // h_t complete before next step's GEMM reads it
        cur ^= 1;
    }

    // ---- partial FC: ws[r][b] = sum_n h_T[b][n] * fc_w[r*H + n] ----
    {
        const _Float16* __restrict__ hfin = hbuf[cur];
        const int row = tid >> 2;             // 0..63
        const int q   = tid & 3;              // quarter of n-range
        float s = 0.f;
        #pragma unroll 4
        for (int n = q * 64; n < q * 64 + 64; ++n)
            s += (float)hfin[row * HPAD + n] * fcwl[n];
        red[tid] = s;
        __syncthreads();
        if (tid < BTILE) {
            const float p = red[tid * 4] + red[tid * 4 + 1]
                          + red[tid * 4 + 2] + red[tid * 4 + 3];
            ws[r * BATCH + b0 + tid] = p;
        }
    }
}

// out[b] = fc_b + sum_r ws[r][b]   (deterministic reduction)
__global__ __launch_bounds__(256) void fc_reduce_kernel(
    const float* __restrict__ ws, const float* __restrict__ fcb,
    float* __restrict__ out)
{
    const int b = blockIdx.x * 256 + threadIdx.x;
    float s = fcb[0];
    #pragma unroll
    for (int r = 0; r < NUM_RNNS; ++r) s += ws[r * BATCH + b];
    out[b] = s;
}

extern "C" void kernel_launch(void* const* d_in, const int* in_sizes, int n_in,
                              void* d_out, int out_size, void* d_ws, size_t ws_size,
                              hipStream_t stream) {
    const float* x    = (const float*)d_in[0];
    const float* Wx   = (const float*)d_in[1];
    const float* Wh   = (const float*)d_in[2];
    const float* bv   = (const float*)d_in[3];
    const float* fcw  = (const float*)d_in[4];
    const float* fcb  = (const float*)d_in[5];
    float*       ws   = (float*)d_ws;          // needs 15*512*4 = 30 KB
    float*       out  = (float*)d_out;         // [512] f32

    rnn_bank_kernel<<<NUM_RNNS * NBT, 256, 0, stream>>>(x, Wx, Wh, bv, fcw, ws);
    fc_reduce_kernel<<<BATCH / 256, 256, 0, stream>>>(ws, fcb, out);
}
